// Encoder_6468220747941
// MI455X (gfx1250) — matrix operand order repaired and checked by reading, not compiled, not run on hardware
//
#include <hip/hip_runtime.h>

// ---------------------------------------------------------------------------
// Transformer encoder (6 layers) for MI455X / gfx1250, wave32 + WMMA f16.
// - All GEMMs via v_wmma_f32_16x16x32_f16; weights pre-transposed to [N,K] f16.
// - Attention fully fused (scores -> masked softmax -> P@V) with the whole
//   16x1024 score row-block stashed in LDS (32KB/wave; CDNA5 has 320KB/WGP),
//   so the 805MB attn-probs output is written to HBM exactly once.
// Workspace: ~97 MB.
// ---------------------------------------------------------------------------

#define DM   512
#define DFF  2048
#define DK   64
#define NH   8
#define NL   6
#define BB   4
#define SS   1024
#define TOKN (BB * SS) // 4096

typedef __attribute__((ext_vector_type(16))) _Float16 v16h;
typedef __attribute__((ext_vector_type(8)))  _Float16 v8h;
typedef __attribute__((ext_vector_type(8)))  float    v8f;

union AFrag { v16h v; v8h h[2]; };

#define WMMA_F16(a, b, c) \
  __builtin_amdgcn_wmma_f32_16x16x32_f16(false, (a), false, (b), (short)0, (c), false, false)

// --------------------------------------------------------------------------
// Generic batched WMMA GEMM:  C = alpha * A[M,K] * Bt[N,K]^T.
// One wave -> 16(M) x 64(N) tile (4 accumulators, A fragment reused 4x).
// --------------------------------------------------------------------------
__device__ __forceinline__ void store_tile(const v8f& c, float* Cf, _Float16* Ch,
                                           int ldcf, int ldch, int m0, int half,
                                           int n, float alpha, int relu) {
#pragma unroll
  for (int v = 0; v < 8; ++v) {
    float val = c[v] * alpha;
    if (relu && val < 0.f) val = 0.f;
    size_t row = (size_t)(m0 + v + 8 * half);
    if (Cf) Cf[row * (size_t)ldcf + n] = val;
    if (Ch) Ch[row * (size_t)ldch + n] = (_Float16)val;
  }
}

__global__ __launch_bounds__(256) void wmma_gemm_bt(
    const _Float16* __restrict__ A, const _Float16* __restrict__ Bt,
    float* __restrict__ Cf, _Float16* __restrict__ Ch,
    int M, int N, int K, int lda, int ldb, int ldcf, int ldch,
    long sAy, long sAz, long sBy, long sBz,
    long sCfy, long sCfz, long sChy, long sChz,
    float alpha, int relu) {
  A  += (size_t)blockIdx.y * sAy + (size_t)blockIdx.z * sAz;
  Bt += (size_t)blockIdx.y * sBy + (size_t)blockIdx.z * sBz;
  if (Cf) Cf += (size_t)blockIdx.y * sCfy + (size_t)blockIdx.z * sCfz;
  if (Ch) Ch += (size_t)blockIdx.y * sChy + (size_t)blockIdx.z * sChz;

  const int lane   = threadIdx.x & 31;
  const int wave   = threadIdx.x >> 5;
  const int tilesN = N >> 6;
  const int wid    = blockIdx.x * 8 + wave;
  if (wid >= (M >> 4) * tilesN) return;
  const int m0   = (wid / tilesN) << 4;
  const int n0   = (wid % tilesN) << 6;
  const int r    = lane & 15;
  const int half = lane >> 4;

  const _Float16* pA = A + (size_t)(m0 + r) * lda + 16 * half;
  const _Float16* pB = Bt + (size_t)(n0 + r) * ldb + 16 * half;
  const size_t bstep = (size_t)16 * ldb;

  v8f c0 = {}, c1 = {}, c2 = {}, c3 = {};
  for (int k0 = 0; k0 < K; k0 += 32) {
    __builtin_prefetch(pB + k0 + 256, 0, 1);  // global_prefetch_b8: next B chunk
    AFrag a;
    a.h[0] = *(const v8h*)(pA + k0);
    a.h[1] = *(const v8h*)(pA + k0 + 8);
    v16h b0 = *(const v16h*)(pB + k0);
    v16h b1 = *(const v16h*)(pB + bstep + k0);
    v16h b2 = *(const v16h*)(pB + 2 * bstep + k0);
    v16h b3 = *(const v16h*)(pB + 3 * bstep + k0);
    c0 = WMMA_F16(a.v, b0, c0);
    c1 = WMMA_F16(a.v, b1, c1);
    c2 = WMMA_F16(a.v, b2, c2);
    c3 = WMMA_F16(a.v, b3, c3);
  }
  store_tile(c0, Cf, Ch, ldcf, ldch, m0, half, n0 + r,      alpha, relu);
  store_tile(c1, Cf, Ch, ldcf, ldch, m0, half, n0 + 16 + r, alpha, relu);
  store_tile(c2, Cf, Ch, ldcf, ldch, m0, half, n0 + 32 + r, alpha, relu);
  store_tile(c3, Cf, Ch, ldcf, ldch, m0, half, n0 + 48 + r, alpha, relu);
}

// ---- weight convert + transpose: dst[l][n*K + k] = (f16) src[l][k*N + n] ----
__global__ __launch_bounds__(256) void transpose_w_kernel(
    const float* __restrict__ src, _Float16* __restrict__ dst,
    int K, int N, long srcStride, long dstStride) {
  long i = (long)blockIdx.x * 256 + threadIdx.x;
  if (i >= (long)K * N) return;
  int n = (int)(i % N), k = (int)(i / N);
  dst[(size_t)blockIdx.y * dstStride + (size_t)n * K + k] =
      (_Float16)src[(size_t)blockIdx.y * srcStride + i];
}

// ---- embedding + positional encoding -> f32 and f16 copies ----
__global__ __launch_bounds__(256) void embed_kernel(
    const int* __restrict__ tok, const float* __restrict__ emb,
    const float* __restrict__ pe, float* __restrict__ x, _Float16* __restrict__ xh) {
  int i = blockIdx.x * 256 + threadIdx.x;
  int t = i >> 9, d = i & (DM - 1);
  int s = t & (SS - 1);
  float v = emb[(size_t)tok[t] * DM + d] + pe[(size_t)s * DM + d];
  x[i]  = v;
  xh[i] = (_Float16)v;
}

// ---- V transpose: vt[b,h,d,s] = qkv[b*S+s, 2*DM + h*64 + d] ----
__global__ __launch_bounds__(256) void transpose_v_kernel(
    const _Float16* __restrict__ qkv, _Float16* __restrict__ vt) {
  int i = blockIdx.x * 256 + threadIdx.x;  // BB*NH*DK*SS, s fastest
  int s = i & (SS - 1);
  int d = (i >> 10) & (DK - 1);
  int h = (i >> 16) & (NH - 1);
  int b = i >> 19;
  vt[i] = qkv[(size_t)(b * SS + s) * (3 * DM) + 2 * DM + h * DK + d];
}

// --------------------------------------------------------------------------
// Fused attention: per wave one 16-row q-tile of one (b,h).
//   scores (WMMA) -> mask+scale -> f16 stash in LDS (16x1024 = 32KB/wave)
//   -> row max (shfl_xor within half) -> e=exp(s-max) stash + row sums
//   -> write normalized probs f32 to d_out once
//   -> ctx = E @ V via WMMA (A-fragments straight from LDS stash), *1/sum.
// 4 waves / block, dynamic LDS = 4*32KB + 1KB scratch.
// --------------------------------------------------------------------------
__global__ __launch_bounds__(128) void fused_attn_kernel(
    const _Float16* __restrict__ qkv, const _Float16* __restrict__ vt,
    float* __restrict__ attnL, _Float16* __restrict__ ctxh,
    const int* __restrict__ tok) {
  extern __shared__ char smem_raw[];
  const int h    = blockIdx.y, b = blockIdx.z;
  const int lane = threadIdx.x & 31;
  const int wave = threadIdx.x >> 5;
  const int r    = lane & 15;
  const int half = lane >> 4;
  const int m0   = (blockIdx.x * 4 + wave) << 4;

  _Float16* sp   = (_Float16*)smem_raw + (size_t)wave * 16 * SS;        // [16][1024]
  float*    sred = (float*)(smem_raw + (size_t)4 * 16 * SS * 2) + wave * 64;

  const _Float16* Qb = qkv + (size_t)b * SS * (3 * DM) + h * DK;
  const _Float16* Kb = Qb + DM;
  const _Float16* Vb = vt + (size_t)(b * NH + h) * DK * SS;             // [64][1024]
  float* P = attnL + (size_t)(b * NH + h) * SS * SS;
  const int* tb = tok + b * SS;

  // mask bits for this lane's columns (col = r + 16*t)
  unsigned long long cmask = 0ull;
  for (int t = 0; t < 64; ++t)
    cmask |= (unsigned long long)(tb[r + 16 * t] == 0) << t;

  // Q fragments for the q-tile (K = 64 -> two k-steps), loaded once
  const _Float16* pQ = Qb + (size_t)(m0 + r) * (3 * DM) + 16 * half;
  AFrag qa0, qa1;
  qa0.h[0] = *(const v8h*)(pQ);
  qa0.h[1] = *(const v8h*)(pQ + 8);
  qa1.h[0] = *(const v8h*)(pQ + 32);
  qa1.h[1] = *(const v8h*)(pQ + 40);

  // ---- pass 1: scores, mask+scale, stash f16, per-lane running row max ----
  v8f mrow;
#pragma unroll
  for (int v = 0; v < 8; ++v) mrow[v] = -3.0e38f;

  for (int n0 = 0; n0 < SS; n0 += 64) {
#pragma unroll
    for (int j = 0; j < 4; ++j) {
      const int col = n0 + 16 * j + r;
      const _Float16* pK = Kb + (size_t)col * (3 * DM) + 16 * half;
      v16h b0 = *(const v16h*)(pK);
      v16h b1 = *(const v16h*)(pK + 32);
      v8f sc = {};
      sc = WMMA_F16(qa0.v, b0, sc);
      sc = WMMA_F16(qa1.v, b1, sc);
      const bool msk = (cmask >> ((n0 >> 4) + j)) & 1;
#pragma unroll
      for (int v = 0; v < 8; ++v) {
        float s = msk ? -30000.f : sc[v] * 0.125f;   // -30000 fits f16; exp->0
        _Float16 sh = (_Float16)s;
        mrow[v] = fmaxf(mrow[v], (float)sh);         // consistent w/ f16 stash
        sp[(size_t)(v + 8 * half) * SS + col] = sh;
      }
    }
  }
  // cross-lane row max within the 16-lane half (rows v+8*half)
#pragma unroll
  for (int off = 1; off < 16; off <<= 1)
#pragma unroll
    for (int v = 0; v < 8; ++v)
      mrow[v] = fmaxf(mrow[v], __shfl_xor(mrow[v], off, 32));
  if (r == 0)
#pragma unroll
    for (int v = 0; v < 8; ++v) sred[8 * half + v] = mrow[v];
  // (same-wave DS ops are in-order; no barrier needed)

  // ---- pass 2a: e = exp(s - max), stash back, partial row sums -----------
  const float rmax    = sred[r];
  const int   rowbase = r * SS + half * 512;   // row r, this half's 512 cols
  float psum = 0.f;
  for (int c = 0; c < 512; c += 8) {
    v8h sv = *(v8h*)(sp + rowbase + c);
    v8h ev;
#pragma unroll
    for (int e = 0; e < 8; ++e) {
      _Float16 xh = (_Float16)__expf((float)sv[e] - rmax);
      ev[e] = xh;
      psum += (float)xh;                        // sum of rounded e -> exact
    }
    *(v8h*)(sp + rowbase + c) = ev;
  }
  sred[16 + half * 16 + r] = psum;
  const float rowsum = sred[16 + r] + sred[32 + r];
  if (half == 0) sred[48 + r] = rowsum;

  // ---- pass 2b: single HBM write of normalized probs ---------------------
  {
    const float inv = 1.0f / rowsum;
    float* prow = P + (size_t)(m0 + r) * SS + half * 512;
    for (int c = 0; c < 512; c += 8) {
      v8h ev = *(v8h*)(sp + rowbase + c);
      float4 o0, o1;
      o0.x = (float)ev[0] * inv; o0.y = (float)ev[1] * inv;
      o0.z = (float)ev[2] * inv; o0.w = (float)ev[3] * inv;
      o1.x = (float)ev[4] * inv; o1.y = (float)ev[5] * inv;
      o1.z = (float)ev[6] * inv; o1.w = (float)ev[7] * inv;
      *(float4*)(prow + c)     = o0;
      *(float4*)(prow + c + 4) = o1;
    }
  }

  // ---- ctx = E @ V (A-fragments from LDS stash), scale by 1/rowsum -------
  v8f a0 = {}, a1 = {}, a2 = {}, a3 = {};
  for (int k0 = 0; k0 < SS; k0 += 32) {
    AFrag pa;
    pa.h[0] = *(v8h*)(sp + (size_t)r * SS + k0 + 16 * half);
    pa.h[1] = *(v8h*)(sp + (size_t)r * SS + k0 + 8 + 16 * half);
    const _Float16* pV = Vb + (size_t)r * SS + k0 + 16 * half;
    v16h b0 = *(const v16h*)(pV);
    v16h b1 = *(const v16h*)(pV + (size_t)16 * SS);
    v16h b2 = *(const v16h*)(pV + (size_t)32 * SS);
    v16h b3 = *(const v16h*)(pV + (size_t)48 * SS);
    a0 = WMMA_F16(pa.v, b0, a0);
    a1 = WMMA_F16(pa.v, b1, a1);
    a2 = WMMA_F16(pa.v, b2, a2);
    a3 = WMMA_F16(pa.v, b3, a3);
  }
  {
    float invr[8];
#pragma unroll
    for (int v = 0; v < 8; ++v) invr[v] = 1.0f / sred[48 + v + 8 * half];
    _Float16* cbase = ctxh + (size_t)(b * SS + m0) * DM + h * DK;
    auto st = [&](const v8f& ac, int j) {
#pragma unroll
      for (int v = 0; v < 8; ++v)
        cbase[(size_t)(v + 8 * half) * DM + 16 * j + r] = (_Float16)(ac[v] * invr[v]);
    };
    st(a0, 0); st(a1, 1); st(a2, 2); st(a3, 3);
  }
}

// ---- residual add + LayerNorm (optional affine); f32 + f16 outputs ----
__global__ __launch_bounds__(256) void add_ln_kernel(
    const float* __restrict__ y, const float* __restrict__ xin,
    const float* __restrict__ g, const float* __restrict__ bb,
    float* __restrict__ xout, _Float16* __restrict__ xh) {
  const int row = blockIdx.x;
  const size_t base = (size_t)row * DM;
  const int tid = threadIdx.x;
  float a0 = y[base + tid]       + xin[base + tid];
  float a1 = y[base + tid + 256] + xin[base + tid + 256];
  __shared__ float s1[256], s2[256];
  s1[tid] = a0 + a1;
  s2[tid] = a0 * a0 + a1 * a1;
  __syncthreads();
  for (int s = 128; s > 0; s >>= 1) {
    if (tid < s) { s1[tid] += s1[tid + s]; s2[tid] += s2[tid + s]; }
    __syncthreads();
  }
  float mean = s1[0] * (1.0f / DM);
  float var  = s2[0] * (1.0f / DM) - mean * mean;
  float rstd = rsqrtf(var + 1e-5f);
  float o0 = (a0 - mean) * rstd;
  float o1 = (a1 - mean) * rstd;
  if (g) { o0 = o0 * g[tid] + bb[tid]; o1 = o1 * g[tid + 256] + bb[tid + 256]; }
  xout[base + tid]       = o0;
  xout[base + tid + 256] = o1;
  xh[base + tid]         = (_Float16)o0;
  xh[base + tid + 256]   = (_Float16)o1;
}

// ---------------------------------------------------------------------------
extern "C" void kernel_launch(void* const* d_in, const int* in_sizes, int n_in,
                              void* d_out, int out_size, void* d_ws, size_t ws_size,
                              hipStream_t stream) {
  const int*   tok  = (const int*)  d_in[0];
  const float* emb  = (const float*)d_in[1];
  const float* pe   = (const float*)d_in[2];
  const float* WQ   = (const float*)d_in[3];
  const float* WK   = (const float*)d_in[4];
  const float* WV   = (const float*)d_in[5];
  const float* WO   = (const float*)d_in[6];
  const float* W1   = (const float*)d_in[7];
  const float* W2   = (const float*)d_in[8];
  const float* ln_g = (const float*)d_in[9];
  const float* ln_b = (const float*)d_in[10];

  float* out_x = (float*)d_out;
  float* attns = out_x + (size_t)TOKN * DM;  // [NL,BB,NH,SS,SS]

  char* wsb = (char*)d_ws;
  size_t off = 0;
  auto alloc = [&](size_t bytes) -> char* {
    char* p = wsb + off;
    off = (off + bytes + 255) & ~(size_t)255;
    return p;
  };
  _Float16* qkvw = (_Float16*)alloc((size_t)NL * 3 * DM * DM * 2);  // [l][1536,512]
  _Float16* wo_t = (_Float16*)alloc((size_t)NL * DM * DM * 2);
  _Float16* w1_t = (_Float16*)alloc((size_t)NL * DFF * DM * 2);
  _Float16* w2_t = (_Float16*)alloc((size_t)NL * DM * DFF * 2);
  float*    x    = (float*)   alloc((size_t)TOKN * DM * 4);
  _Float16* xh   = (_Float16*)alloc((size_t)TOKN * DM * 2);
  _Float16* qkv  = (_Float16*)alloc((size_t)TOKN * 3 * DM * 2);
  _Float16* vt   = (_Float16*)alloc((size_t)BB * NH * DK * SS * 2);
  _Float16* ctxh = (_Float16*)alloc((size_t)TOKN * DM * 2);
  float*    yb   = (float*)   alloc((size_t)TOKN * DM * 4);
  _Float16* hh   = (_Float16*)alloc((size_t)TOKN * DFF * 2);
  (void)off; (void)ws_size;

  // weights -> transposed f16 (QKV packed per layer as [1536,512])
  {
    dim3 g((DM * DM + 255) / 256, NL);
    transpose_w_kernel<<<g, 256, 0, stream>>>(WQ, qkvw,               DM, DM, (long)DM * DM, (long)3 * DM * DM);
    transpose_w_kernel<<<g, 256, 0, stream>>>(WK, qkvw + DM * DM,     DM, DM, (long)DM * DM, (long)3 * DM * DM);
    transpose_w_kernel<<<g, 256, 0, stream>>>(WV, qkvw + 2 * DM * DM, DM, DM, (long)DM * DM, (long)3 * DM * DM);
    transpose_w_kernel<<<g, 256, 0, stream>>>(WO, wo_t,               DM, DM, (long)DM * DM, (long)DM * DM);
  }
  {
    dim3 g((DM * DFF + 255) / 256, NL);
    transpose_w_kernel<<<g, 256, 0, stream>>>(W1, w1_t, DM, DFF, (long)DM * DFF, (long)DM * DFF);
    transpose_w_kernel<<<g, 256, 0, stream>>>(W2, w2_t, DFF, DM, (long)DFF * DM, (long)DFF * DM);
  }

  embed_kernel<<<TOKN * DM / 256, 256, 0, stream>>>(tok, emb, pe, x, xh);

  auto gemm = [&](const _Float16* A, const _Float16* Bt, float* Cf, _Float16* Ch,
                  int M, int N, int K, int lda, int ldb, int ldcf, int ldch,
                  float alpha, int relu) {
    dim3 grid(((M >> 4) * (N >> 6) + 7) / 8, 1, 1);
    wmma_gemm_bt<<<grid, 256, 0, stream>>>(A, Bt, Cf, Ch, M, N, K, lda, ldb, ldcf, ldch,
                                           0, 0, 0, 0, 0, 0, 0, 0, alpha, relu);
  };

  const int attn_lds = 4 * 16 * SS * 2 + 4 * 64 * 4;  // 132096 bytes

  for (int l = 0; l < NL; ++l) {
    // fused QKV projection: [4096,512] x [512,1536] -> f16 [4096,1536]
    gemm(xh, qkvw + (size_t)l * 3 * DM * DM, nullptr, qkv,
         TOKN, 3 * DM, DM, DM, DM, 0, 3 * DM, 1.0f, 0);

    transpose_v_kernel<<<(BB * NH * DK * SS) / 256, 256, 0, stream>>>(qkv, vt);

    // fused attention: scores -> masked softmax (probs to d_out) -> ctx
    float* attnL = attns + (size_t)l * BB * NH * SS * SS;
    fused_attn_kernel<<<dim3(SS / 64, NH, BB), 128, attn_lds, stream>>>(
        qkv, vt, attnL, ctxh, tok);

    // O = ctx @ WO -> f32 ; LN(O + x)
    gemm(ctxh, wo_t + (size_t)l * DM * DM, yb, nullptr,
         TOKN, DM, DM, DM, DM, DM, 0, 1.0f, 0);
    add_ln_kernel<<<TOKN, 256, 0, stream>>>(yb, x, nullptr, nullptr, x, xh);

    // FFN: H = relu(x @ W1) -> f16 ; Y = H @ W2 -> f32 ; LN(Y + x)
    gemm(xh, w1_t + (size_t)l * DFF * DM, nullptr, hh,
         TOKN, DFF, DM, DM, DM, 0, DFF, 1.0f, 1);
    gemm(hh, w2_t + (size_t)l * DM * DFF, yb, nullptr,
         TOKN, DM, DFF, DFF, DFF, DM, 0, 1.0f, 0);
    float* xo = (l == NL - 1) ? out_x : x;
    add_ln_kernel<<<TOKN, 256, 0, stream>>>(yb, x, ln_g + (size_t)l * DM, ln_b + (size_t)l * DM, xo, xh);
  }
}
